// SASREC_90967407329640
// MI455X (gfx1250) — compile-verified, matrix operand order checked by reading, not run on hardware
//
#include <hip/hip_runtime.h>

// ---------------------------------------------------------------------------
// SASREC-style 2-layer causal transformer encoder for gfx1250 (MI455X).
// GEMMs -> v_wmma_f32_16x16x32_f16; weights staged to LDS via the Tensor
// Data Mover (tensor_load_to_lds, TENSORcnt) and shared by 8 waves per WG.
// ---------------------------------------------------------------------------

typedef __attribute__((ext_vector_type(16))) _Float16     v16h;
typedef __attribute__((ext_vector_type(8)))  _Float16     v8h;
typedef __attribute__((ext_vector_type(8)))  float        v8f;
typedef __attribute__((ext_vector_type(4)))  unsigned int v4u;
typedef __attribute__((ext_vector_type(4)))  int          v4i;
typedef __attribute__((ext_vector_type(8)))  int          v8i;

#define LNUM 2
#define HNUM 2
#define DDIM 128
#define SLEN 200
#define BSZ  512
#define MTOK (BSZ * SLEN)          // 102400 tokens (multiple of 16)
#define SP   224                   // padded seq len: 7 k-steps of 32
#define DH   64                    // head dim
#define WELEM (DDIM * DDIM)        // 16384 elements per weight matrix
#define NEGV (-4294967295.0f)      // -2^32 + 1 (rounds same as reference f32)

// ---------------------------------------------------------------------------
// Fragment helpers (CDNA5 16x16x32 f16 WMMA layouts).
// A-fragment: lane L<16 holds row L, K={0..7} in VGPRs0-3, K={16..23} in
// VGPRs4-7; lanes 16-31 hold the K+8 chunks. B symmetric with N-major data.
// ---------------------------------------------------------------------------
__device__ __forceinline__ v16h load_frag(const _Float16* p0, int stride) {
    int lane = threadIdx.x & 31;
    const _Float16* p = p0 + (size_t)(lane & 15) * stride + ((lane >> 4) << 3);
    union { v16h v; v8h h[2]; } u;
    u.h[0] = *(const v8h*)p;          // K = ko .. ko+7
    u.h[1] = *(const v8h*)(p + 16);   // K = ko+16 .. ko+23
    return u.v;
}

__device__ __forceinline__ v16h load_frag_f32(const float* p0, int stride) {
    int lane = threadIdx.x & 31;
    const float* p = p0 + (size_t)(lane & 15) * stride + ((lane >> 4) << 3);
    v16h r;
#pragma unroll
    for (int i = 0; i < 8; ++i) r[i] = (_Float16)p[i];
#pragma unroll
    for (int i = 0; i < 8; ++i) r[8 + i] = (_Float16)p[16 + i];
    return r;
}

__device__ __forceinline__ v8f wmma32(v16h a, v16h b, v8f c) {
    return __builtin_amdgcn_wmma_f32_16x16x32_f16(
        false, a, false, b, (short)0, c, false, false);
}

// C/D fragment: VGPR r -> row = r + 8*(lane>>4), col = lane & 15.

// ---------------------------------------------------------------------------
// Tensor Data Mover: 1-D copy of `nelem` f16 (nelem <= 65535) global -> LDS.
// D# per CDNA5 ISA ch.8: group0 {count=1, lds_addr, global_addr[56:0],
// type=2}; group1 {wg_mask=0, data_size=1(2B), tensor_dim0=tile_dim0=nelem,
// tensor_dim1=tile_dim1=1, dim0_stride=nelem}. Groups 2/3 zero (<=2D).
// This toolchain (clang-23 / therock-10.0) exposes the 6-arg builtin:
//   (u32x4 g0, i32x8 g1, i32x4 g2, i32x4 g3, i32x8 extra, i32 cpol)
// ---------------------------------------------------------------------------
__device__ __forceinline__ void tdm_load_lds(const _Float16* gsrc,
                                             _Float16* ldst, unsigned nelem) {
#if __has_builtin(__builtin_amdgcn_tensor_load_to_lds)
    unsigned long long ga = (unsigned long long)(size_t)gsrc;
    unsigned lp = (unsigned)(size_t)ldst;            // addr[31:0] == LDS offset
    v4u g0;
    g0[0] = 1u;                                      // count=1 (valid user D#)
    g0[1] = lp;                                      // lds_addr
    g0[2] = (unsigned)(ga & 0xffffffffu);            // global_addr[31:0]
    g0[3] = (unsigned)((ga >> 32) & 0x01ffffffu) | (2u << 30); // [56:32]|type=2
    v8i g1;
    g1[0] = (int)(1u << 16);                         // data_size=1 -> 2 bytes
    g1[1] = (int)((nelem & 0xffffu) << 16);          // tensor_dim0[15:0]
    g1[2] = (int)(((nelem >> 16) & 0xffffu) | (1u << 16)); // dim0[31:16],dim1=1
    g1[3] = (int)((nelem & 0xffffu) << 16);          // tile_dim0 = nelem
    g1[4] = 1;                                       // tile_dim1 = 1
    g1[5] = (int)nelem;                              // dim0_stride[31:0]
    g1[6] = 0;
    g1[7] = 0;
    v4i gz4 = {0, 0, 0, 0};
    v8i gz8 = {0, 0, 0, 0, 0, 0, 0, 0};
    __builtin_amdgcn_tensor_load_to_lds(g0, g1, gz4, gz4, gz8, 0);
#else
    for (unsigned i = threadIdx.x & 31; i < nelem; i += 32) ldst[i] = gsrc[i];
#endif
}

__device__ __forceinline__ void tdm_wait() {
#if __has_builtin(__builtin_amdgcn_s_wait_tensorcnt)
    __builtin_amdgcn_s_wait_tensorcnt(0);
#endif
}

// ---------------------------------------------------------------------------
// Weight prep: Wt[n][k] = (f16) W[k][n]  (N-major f16 for B-fragments)
// ---------------------------------------------------------------------------
__global__ void wprep_kernel(const float* __restrict__ W, _Float16* __restrict__ Wt) {
    int i = blockIdx.x * blockDim.x + threadIdx.x;   // i = n*D + k
    if (i < WELEM) {
        int n = i / DDIM, k = i % DDIM;
        Wt[i] = (_Float16)W[k * DDIM + n];
    }
}

// ---------------------------------------------------------------------------
// LayerNorm + mask extraction. One block (128 threads) per token.
// ---------------------------------------------------------------------------
__global__ void ln_kernel(const float* __restrict__ x,
                          const float* __restrict__ g, const float* __restrict__ bb,
                          _Float16* __restrict__ xh, _Float16* __restrict__ xnh,
                          float* __restrict__ km, float* __restrict__ qm) {
    __shared__ float red[DDIM];
    int t = blockIdx.x, d = threadIdx.x;
    int s = t % SLEN;
    float v = x[(size_t)t * DDIM + d];
    xh[(size_t)t * DDIM + d] = (_Float16)v;

    red[d] = v; __syncthreads();
    for (int o = 64; o > 0; o >>= 1) { if (d < o) red[d] += red[d + o]; __syncthreads(); }
    float sum = red[0]; __syncthreads();
    float mean = sum * (1.0f / DDIM);
    float dv = v - mean;

    red[d] = dv * dv; __syncthreads();
    for (int o = 64; o > 0; o >>= 1) { if (d < o) red[d] += red[d + o]; __syncthreads(); }
    float var = red[0] * (1.0f / DDIM); __syncthreads();

    float xn = g[s * DDIM + d] * dv * rsqrtf(var + 1e-8f) + bb[s * DDIM + d];
    xnh[(size_t)t * DDIM + d] = (_Float16)xn;

    red[d] = xn; __syncthreads();
    for (int o = 64; o > 0; o >>= 1) { if (d < o) red[d] += red[d + o]; __syncthreads(); }
    if (d == 0) {
        km[t] = (sum != 0.0f) ? 1.0f : 0.0f;       // sign(|sum(x)|)
        qm[t] = (red[0] != 0.0f) ? 1.0f : 0.0f;    // sign(|sum(x_norm)|)
    }
}

// ---------------------------------------------------------------------------
// QKV GEMMs: 8 waves / WG, each wave one 16-token tile; Wq/Wk/Wv staged to
// LDS once per WG via TDM.  q scaled by 0.125; q,k -> [B,H,SP,DH] row-major,
// v -> [B,H,DH,SP] (transposed-B form for probs@v).
// ---------------------------------------------------------------------------
__global__ void __launch_bounds__(256) qkv_kernel(
    const _Float16* __restrict__ xnh, const _Float16* __restrict__ xh,
    const _Float16* __restrict__ wqT, const _Float16* __restrict__ wkT,
    const _Float16* __restrict__ wvT,
    const float* __restrict__ bq, const float* __restrict__ bk,
    const float* __restrict__ bv,
    _Float16* __restrict__ qO, _Float16* __restrict__ kO, _Float16* __restrict__ vO) {
    extern __shared__ __align__(16) _Float16 smemq[];
    _Float16* wq_s = smemq;
    _Float16* wk_s = smemq + WELEM;
    _Float16* wv_s = smemq + 2 * WELEM;
    if (threadIdx.x < 32) {
        tdm_load_lds(wqT, wq_s, WELEM);
        tdm_load_lds(wkT, wk_s, WELEM);
        tdm_load_lds(wvT, wv_s, WELEM);
        tdm_wait();
    }

    int wave = threadIdx.x >> 5, lane = threadIdx.x & 31;
    int row0 = (blockIdx.x * 8 + wave) * 16;
    int col = lane & 15, rbase = (lane >> 4) << 3;

    v16h aq[4], ax[4];
#pragma unroll
    for (int ks = 0; ks < 4; ++ks) {
        aq[ks] = load_frag(xnh + (size_t)row0 * DDIM + ks * 32, DDIM);
        ax[ks] = load_frag(xh  + (size_t)row0 * DDIM + ks * 32, DDIM);
    }
    __syncthreads();

    for (int nt = 0; nt < 8; ++nt) {
        v8f cq = {}, ck = {}, cv = {};
#pragma unroll
        for (int ks = 0; ks < 4; ++ks) {
            cq = wmma32(aq[ks], load_frag(wq_s + nt * 16 * DDIM + ks * 32, DDIM), cq);
            ck = wmma32(ax[ks], load_frag(wk_s + nt * 16 * DDIM + ks * 32, DDIM), ck);
            cv = wmma32(ax[ks], load_frag(wv_s + nt * 16 * DDIM + ks * 32, DDIM), cv);
        }
        int n = nt * 16 + col;
        float biasq = bq[n], biask = bk[n], biasv = bv[n];
        int h = n >> 6, di = n & 63;
#pragma unroll
        for (int r = 0; r < 8; ++r) {
            int t = row0 + r + rbase;
            int b = t / SLEN, s = t % SLEN;
            size_t bh = (size_t)(b * HNUM + h);
            qO[(bh * SP + s) * DH + di] = (_Float16)(0.125f * (cq[r] + biasq));
            kO[(bh * SP + s) * DH + di] = (_Float16)(ck[r] + biask);
            vO[(bh * DH + di) * SP + s] = (_Float16)(cv[r] + biasv);
        }
    }
}

// ---------------------------------------------------------------------------
// Attention: one wave per (b, h, 16-query tile).  Scores via WMMA, masks +
// softmax in LDS, probs@v via WMMA, +x_norm residual -> y.
// ---------------------------------------------------------------------------
__global__ void __launch_bounds__(32) attn_kernel(
    const _Float16* __restrict__ qI, const _Float16* __restrict__ kI,
    const _Float16* __restrict__ vT,
    const float* __restrict__ km, const float* __restrict__ qm,
    const _Float16* __restrict__ xnh, float* __restrict__ y) {
    __shared__ __align__(16) float    sc[16][SP];
    __shared__ __align__(16) _Float16 pr[16][SP];
    int lane = threadIdx.x;
    int qt = blockIdx.x, h = blockIdx.y, b = blockIdx.z;
    int row0 = qt * 16;
    int col = lane & 15, rbase = (lane >> 4) << 3;
    size_t bh = (size_t)(b * HNUM + h);

    const _Float16* qbase = qI + (bh * SP + row0) * DH;
    v16h aq0 = load_frag(qbase, DH);
    v16h aq1 = load_frag(qbase + 32, DH);

    // scores = (q*0.125) @ k^T   (k row-major == transposed-B form)
    for (int kt = 0; kt < 13; ++kt) {
        v8f c = {};
        const _Float16* kb = kI + (bh * SP + kt * 16) * DH;
        c = wmma32(aq0, load_frag(kb, DH), c);
        c = wmma32(aq1, load_frag(kb + 32, DH), c);
        int kidx = kt * 16 + col;
        float kmv = (kidx < SLEN) ? km[b * SLEN + kidx] : 0.0f;
#pragma unroll
        for (int r = 0; r < 8; ++r) {
            int qs = row0 + r + rbase;
            float v = c[r];
            if (kidx >= SLEN || kmv == 0.0f || kidx > qs) v = NEGV;
            sc[r + rbase][kidx] = v;
        }
    }
    __syncthreads();

    // per-row softmax (rows 0..15 handled by lanes 0..15)
    if (lane < 16) {
        int qs = row0 + lane;
        if (qs < SLEN) {
            float mx = -3.4e38f;
            for (int k2 = 0; k2 < SLEN; ++k2) mx = fmaxf(mx, sc[lane][k2]);
            float sumv = 0.0f;
            for (int k2 = 0; k2 < SLEN; ++k2) sumv += __expf(sc[lane][k2] - mx);
            float inv = qm[b * SLEN + qs] / sumv;
            for (int k2 = 0; k2 < SLEN; ++k2)
                pr[lane][k2] = (_Float16)(__expf(sc[lane][k2] - mx) * inv);
            for (int k2 = SLEN; k2 < SP; ++k2) pr[lane][k2] = (_Float16)0.0f;
        } else {
            for (int k2 = 0; k2 < SP; ++k2) pr[lane][k2] = (_Float16)0.0f;
        }
    }
    __syncthreads();

    // out = probs @ v   (v_t[dh][SP] row-major == transposed-B form)
    for (int nt = 0; nt < 4; ++nt) {
        v8f c = {};
        const _Float16* vb = vT + (bh * DH + nt * 16) * SP;
#pragma unroll
        for (int ks = 0; ks < 7; ++ks) {
            v16h ap = load_frag(&pr[0][0] + ks * 32, SP);
            v16h bp = load_frag(vb + ks * 32, SP);
            c = wmma32(ap, bp, c);
        }
#pragma unroll
        for (int r = 0; r < 8; ++r) {
            int qs = row0 + r + rbase;
            if (qs < SLEN) {
                size_t t = (size_t)b * SLEN + qs;
                int cg = h * DH + nt * 16 + col;
                y[t * DDIM + cg] = c[r] + (float)xnh[t * DDIM + cg];
            }
        }
    }
}

// ---------------------------------------------------------------------------
// FFN: h = relu(y@W1+b1)@W2 + b2 + y; out = h * mask.  8 waves / WG; W1,W2
// staged to LDS via TDM; per-wave ReLU intermediate in LDS.
// ---------------------------------------------------------------------------
__global__ void __launch_bounds__(256) ffn_kernel(
    const float* __restrict__ y,
    const _Float16* __restrict__ w1T, const _Float16* __restrict__ w2T,
    const float* __restrict__ b1, const float* __restrict__ b2,
    const float* __restrict__ mask, float* __restrict__ xout) {
    extern __shared__ __align__(16) _Float16 smemf[];
    _Float16* w1_s = smemf;
    _Float16* w2_s = smemf + WELEM;
    if (threadIdx.x < 32) {
        tdm_load_lds(w1T, w1_s, WELEM);
        tdm_load_lds(w2T, w2_s, WELEM);
        tdm_wait();
    }
    int wave = threadIdx.x >> 5, lane = threadIdx.x & 31;
    _Float16* h1 = smemf + 2 * WELEM + wave * (16 * DDIM);
    int row0 = (blockIdx.x * 8 + wave) * 16;
    int col = lane & 15, rbase = (lane >> 4) << 3;

    v16h a[4];
#pragma unroll
    for (int ks = 0; ks < 4; ++ks)
        a[ks] = load_frag_f32(y + (size_t)row0 * DDIM + ks * 32, DDIM);
    __syncthreads();

    for (int nt = 0; nt < 8; ++nt) {
        v8f c = {};
#pragma unroll
        for (int ks = 0; ks < 4; ++ks)
            c = wmma32(a[ks], load_frag(w1_s + nt * 16 * DDIM + ks * 32, DDIM), c);
        int n = nt * 16 + col;
        float bias = b1[n];
#pragma unroll
        for (int r = 0; r < 8; ++r)
            h1[(r + rbase) * DDIM + n] = (_Float16)fmaxf(c[r] + bias, 0.0f);
    }
    __syncthreads();

    v16h a2[4];
#pragma unroll
    for (int ks = 0; ks < 4; ++ks)
        a2[ks] = load_frag(h1 + ks * 32, DDIM);

    for (int nt = 0; nt < 8; ++nt) {
        v8f c = {};
#pragma unroll
        for (int ks = 0; ks < 4; ++ks)
            c = wmma32(a2[ks], load_frag(w2_s + nt * 16 * DDIM + ks * 32, DDIM), c);
        int n = nt * 16 + col;
        float bias = b2[n];
#pragma unroll
        for (int r = 0; r < 8; ++r) {
            int t = row0 + r + rbase;
            xout[(size_t)t * DDIM + n] =
                (c[r] + bias + y[(size_t)t * DDIM + n]) * mask[t];
        }
    }
}

// ---------------------------------------------------------------------------
// Host orchestration (graph-capture safe: memsets + launches on `stream`).
// ---------------------------------------------------------------------------
extern "C" void kernel_launch(void* const* d_in, const int* in_sizes, int n_in,
                              void* d_out, int out_size, void* d_ws, size_t ws_size,
                              hipStream_t stream) {
    (void)in_sizes; (void)n_in; (void)out_size; (void)ws_size;
    const float* x    = (const float*)d_in[0];
    const float* mask = (const float*)d_in[1];
    const float* Wq   = (const float*)d_in[2];
    const float* bq   = (const float*)d_in[3];
    const float* Wk   = (const float*)d_in[4];
    const float* bk   = (const float*)d_in[5];
    const float* Wv   = (const float*)d_in[6];
    const float* bv   = (const float*)d_in[7];
    const float* lng  = (const float*)d_in[8];
    const float* lnb  = (const float*)d_in[9];
    const float* W1   = (const float*)d_in[10];
    const float* b1   = (const float*)d_in[11];
    const float* W2   = (const float*)d_in[12];
    const float* b2   = (const float*)d_in[13];

    char* ws = (char*)d_ws;
    size_t off = 0;
    auto walloc = [&](size_t bytes) -> void* {
        void* p = ws + off;
        off += (bytes + 255) & ~(size_t)255;
        return p;
    };

    const size_t qkvElems = (size_t)BSZ * HNUM * SP * DH;           // padded
    _Float16* wt   = (_Float16*)walloc((size_t)5 * LNUM * WELEM * sizeof(_Float16));
    _Float16* xh   = (_Float16*)walloc((size_t)MTOK * DDIM * sizeof(_Float16));
    _Float16* xnh  = (_Float16*)walloc((size_t)MTOK * DDIM * sizeof(_Float16));
    float*    kmB  = (float*)   walloc((size_t)MTOK * sizeof(float));
    float*    qmB  = (float*)   walloc((size_t)MTOK * sizeof(float));
    _Float16* qB   = (_Float16*)walloc(qkvElems * sizeof(_Float16));
    _Float16* kB   = (_Float16*)walloc(qkvElems * sizeof(_Float16));
    _Float16* vB   = (_Float16*)walloc(qkvElems * sizeof(_Float16));
    float*    yB   = (float*)   walloc((size_t)MTOK * DDIM * sizeof(float));
    float*    x1   = (float*)   walloc((size_t)MTOK * DDIM * sizeof(float));

    // zero q/k/v so the SP-padding region contributes exact zeros
    (void)hipMemsetAsync(qB, 0, qkvElems * sizeof(_Float16), stream);
    (void)hipMemsetAsync(kB, 0, qkvElems * sizeof(_Float16), stream);
    (void)hipMemsetAsync(vB, 0, qkvElems * sizeof(_Float16), stream);

    // weight prep: transposed f16 copies, layout [matrix][layer][D][D]
    const float* weights5[5] = {Wq, Wk, Wv, W1, W2};
    for (int m = 0; m < 5; ++m)
        for (int l = 0; l < LNUM; ++l)
            wprep_kernel<<<(WELEM + 255) / 256, 256, 0, stream>>>(
                weights5[m] + (size_t)l * WELEM,
                wt + ((size_t)m * LNUM + l) * WELEM);
    auto WT = [&](int m, int l) { return wt + ((size_t)m * LNUM + l) * WELEM; };

    const int nWG = MTOK / 16 / 8;            // 800 workgroups of 8 waves
    const size_t qkvLds = (size_t)3 * WELEM * sizeof(_Float16);            // 96 KB
    const size_t ffnLds = (size_t)(2 * WELEM + 8 * 16 * DDIM) * sizeof(_Float16); // 96 KB

    for (int l = 0; l < LNUM; ++l) {
        const float* xin  = (l == 0) ? x : x1;
        float*       xout = (l == LNUM - 1) ? (float*)d_out : x1;

        ln_kernel<<<MTOK, DDIM, 0, stream>>>(
            xin, lng + (size_t)l * SLEN * DDIM, lnb + (size_t)l * SLEN * DDIM,
            xh, xnh, kmB, qmB);

        qkv_kernel<<<nWG, 256, qkvLds, stream>>>(
            xnh, xh, WT(0, l), WT(1, l), WT(2, l),
            bq + l * DDIM, bk + l * DDIM, bv + l * DDIM, qB, kB, vB);

        attn_kernel<<<dim3(13, HNUM, BSZ), 32, 0, stream>>>(
            qB, kB, vB, kmB, qmB, xnh, yB);

        ffn_kernel<<<nWG, 256, ffnLds, stream>>>(
            yB, WT(3, l), WT(4, l), b1 + l * DDIM, b2 + l * DDIM, mask, xout);
    }
}
